// UNetSphericalHealpix_5231270166893
// MI455X (gfx1250) — compile-verified, matrix operand order checked by reading, not run on hardware
//
#include <hip/hip_runtime.h>
#include <hip/hip_bf16.h>

// ---------------------------------------------------------------------------
// Types for CDNA5 WMMA (wave32, 16x16x32 bf16 -> f32)
// ---------------------------------------------------------------------------
typedef __attribute__((ext_vector_type(16))) __bf16 v16bf;
typedef __attribute__((ext_vector_type(8)))  float  v8f;

#define BATCH 16
#define DEG   20

__device__ __forceinline__ unsigned short f2bf(float f) {
    union { float f; unsigned int u; } c; c.f = f;
    unsigned int u = c.u;
    if ((u & 0x7fffffffu) > 0x7f800000u) return (unsigned short)((u >> 16) | 0x40); // NaN
    unsigned int r = u + 0x7fffu + ((u >> 16) & 1u); // round-to-nearest-even
    return (unsigned short)(r >> 16);
}

// ---------------------------------------------------------------------------
// Layout: activations are [V, B, F] fp32 (row r = v*B + b).
// ---------------------------------------------------------------------------

// x [B,V,F] -> act [V,B,F]
__global__ void k_transpose_in(float* __restrict__ act, const float* __restrict__ x,
                               int V, int F) {
    int t = blockIdx.x * blockDim.x + threadIdx.x;
    int n = V * BATCH * F;
    if (t >= n) return;
    int f = t % F; int vb = t / F; int b = vb % BATCH; int v = vb / BATCH;
    act[t] = x[((size_t)b * V + v) * F + f];
}

// Chebyshev pass 1: T1 = L*src  (edge-list gather; edges of node v at v*DEG..)
// also writes bf16 slots 0 (=src) and 1 (=T1) of Xk[M, Kpad]
__global__ void k_cheb1(unsigned short* __restrict__ Xk, float* __restrict__ T1,
                        const float* __restrict__ src, const int* __restrict__ col,
                        const float* __restrict__ val, int V, int fin, int Kpad) {
    int t = blockIdx.x * blockDim.x + threadIdx.x;
    int m2 = BATCH * fin;
    if (t >= V * m2) return;
    int v = t / m2, m = t % m2;
    int e0 = v * DEG;
    float acc = 0.f;
#pragma unroll 4
    for (int e = 0; e < DEG; ++e)
        acc += val[e0 + e] * src[(size_t)col[e0 + e] * m2 + m];
    T1[t] = acc;
    int b = m / fin, f = m % fin;
    size_t r = (size_t)(v * BATCH + b) * Kpad;
    Xk[r + f]        = f2bf(src[t]);
    Xk[r + fin + f]  = f2bf(acc);
}

// Chebyshev pass 2: t2 = 2*L*T1 - src -> bf16 slot 2 of Xk
__global__ void k_cheb2(unsigned short* __restrict__ Xk, const float* __restrict__ T1,
                        const float* __restrict__ src, const int* __restrict__ col,
                        const float* __restrict__ val, int V, int fin, int Kpad) {
    int t = blockIdx.x * blockDim.x + threadIdx.x;
    int m2 = BATCH * fin;
    if (t >= V * m2) return;
    int v = t / m2, m = t % m2;
    int e0 = v * DEG;
    float acc = 0.f;
#pragma unroll 4
    for (int e = 0; e < DEG; ++e)
        acc += val[e0 + e] * T1[(size_t)col[e0 + e] * m2 + m];
    float t2 = 2.f * acc - src[t];
    int b = m / fin, f = m % fin;
    size_t r = (size_t)(v * BATCH + b) * Kpad;
    Xk[r + 2 * fin + f] = f2bf(t2);
}

// zero padded tail columns [Kd, Kpad) of Xk
__global__ void k_zpad(unsigned short* __restrict__ Xk, int Kd, int Kpad, int M) {
    int pad = Kpad - Kd;
    int t = blockIdx.x * blockDim.x + threadIdx.x;
    if (t >= M * pad) return;
    int r = t / pad, c = Kd + t % pad;
    Xk[(size_t)r * Kpad + c] = 0;
}

// w [K=3, fin, fout] fp32  ->  Wt [Npad, Kpad] bf16, Wt[n][k*fin+f] = w[k][f][n]
__global__ void k_prepw(unsigned short* __restrict__ Wt, const float* __restrict__ w,
                        int fin, int fout, int Kd, int Kpad, int Npad) {
    int t = blockIdx.x * blockDim.x + threadIdx.x;
    if (t >= Npad * Kpad) return;
    int n = t / Kpad, c = t % Kpad;
    float v = 0.f;
    if (n < fout && c < Kd) {
        int k = c / fin, f = c % fin;
        v = w[((size_t)k * fin + f) * fout + n];
    }
    Wt[t] = f2bf(v);
}

// WMMA GEMM: dst[M, fout] = Xk[M, Kpad](bf16) * Wt[Npad, Kpad]^T(bf16) + bias
// one wave per 16x16 tile; 4 waves per block.
__global__ void k_gemm(float* __restrict__ dst, const unsigned short* __restrict__ Xk,
                       const unsigned short* __restrict__ Wt, const float* __restrict__ bias,
                       int Kpad, int fout, int tilesM, int tilesN) {
    int tile = blockIdx.x * (blockDim.x >> 5) + (threadIdx.x >> 5);
    if (tile >= tilesM * tilesN) return;               // wave-uniform
    int mt = tile % tilesM, nt = tile / tilesM;
    int lane = threadIdx.x & 31;
    int half = lane >> 4, lr = lane & 15;
    // A: lane (half,lr) holds row mt*16+lr, K elements {half*8..+7, 16+half*8..+7}
    const unsigned short* arow = Xk + (size_t)(mt * 16 + lr) * Kpad + half * 8;
    // B: lane holds column n = nt*16+lr of W (row of Wt), same K striping
    const unsigned short* brow = Wt + (size_t)(nt * 16 + lr) * Kpad + half * 8;

    v8f acc = {};
    for (int kk = 0; kk < Kpad; kk += 32) {
        union { uint4 q[2]; v16bf v; } a, b;
        a.q[0] = *(const uint4*)(arow + kk);
        a.q[1] = *(const uint4*)(arow + kk + 16);
        b.q[0] = *(const uint4*)(brow + kk);
        b.q[1] = *(const uint4*)(brow + kk + 16);
        __builtin_prefetch((const void*)(arow + kk + 64), 0, 1);
        __builtin_prefetch((const void*)(brow + kk + 64), 0, 1);
        acc = __builtin_amdgcn_wmma_f32_16x16x32_bf16(
            /*neg_a=*/false, a.v, /*neg_b=*/false, b.v,
            /*c_mod=*/(short)0, acc, /*reuse_a=*/false, /*reuse_b=*/false);
    }
    int n = nt * 16 + lr;
    if (n < fout) {
        float bv = bias[n];
        int mbase = mt * 16 + half * 8;               // C/D: VGPR r -> M = r + half*8
#pragma unroll
        for (int r = 0; r < 8; ++r)
            dst[(size_t)(mbase + r) * fout + n] = acc[r] + bv;
    }
}

// per-channel mean / rsqrt(var+eps) over all M rows
__global__ void k_stats(float* __restrict__ stats, const float* __restrict__ y,
                        int M, int fout) {
    __shared__ float ss[256], sq[256];
    int f = blockIdx.x;
    float s = 0.f, q = 0.f;
    for (int i = threadIdx.x; i < M; i += blockDim.x) {
        float v = y[(size_t)i * fout + f];
        s += v; q += v * v;
    }
    ss[threadIdx.x] = s; sq[threadIdx.x] = q;
    __syncthreads();
    for (int w = 128; w > 0; w >>= 1) {
        if (threadIdx.x < (unsigned)w) {
            ss[threadIdx.x] += ss[threadIdx.x + w];
            sq[threadIdx.x] += sq[threadIdx.x + w];
        }
        __syncthreads();
    }
    if (threadIdx.x == 0) {
        float m = ss[0] / (float)M;
        float var = sq[0] / (float)M - m * m;
        if (var < 0.f) var = 0.f;
        stats[2 * f]     = m;
        stats[2 * f + 1] = rsqrtf(var + 1e-5f);
    }
}

// in-place BN + ReLU
__global__ void k_bnrelu(float* __restrict__ y, const float* __restrict__ stats,
                         const float* __restrict__ gamma, const float* __restrict__ beta,
                         int M, int fout) {
    int t = blockIdx.x * blockDim.x + threadIdx.x;
    if (t >= M * fout) return;
    int f = t % fout;
    float v = (y[t] - stats[2 * f]) * stats[2 * f + 1] * gamma[f] + beta[f];
    y[t] = v > 0.f ? v : 0.f;
}

// max-pool groups of 4 nodes; idx = argmax (first max)
__global__ void k_pool(float* __restrict__ dst, int* __restrict__ idx,
                       const float* __restrict__ src, int Vp, int F) {
    int t = blockIdx.x * blockDim.x + threadIdx.x;
    if (t >= Vp * BATCH * F) return;
    int f = t % F; int pb = t / F; int b = pb % BATCH; int p = pb / BATCH;
    float best = src[((size_t)(4 * p) * BATCH + b) * F + f];
    int bj = 0;
#pragma unroll
    for (int j = 1; j < 4; ++j) {
        float v = src[((size_t)(4 * p + j) * BATCH + b) * F + f];
        if (v > best) { best = v; bj = j; }
    }
    dst[t] = best;
    idx[t] = bj;
}

// unpool into concat dst (channel offset off, total channels Fdst)
__global__ void k_unpool_cat(float* __restrict__ dst, int Fdst, int off,
                             const float* __restrict__ src, const int* __restrict__ idx,
                             int Vp, int F) {
    int t = blockIdx.x * blockDim.x + threadIdx.x;
    int n = Vp * 4 * BATCH * F;
    if (t >= n) return;
    int f = t % F; int vb = t / F; int b = vb % BATCH; int v = vb / BATCH;
    int p = v >> 2, j = v & 3;
    size_t sidx = ((size_t)p * BATCH + b) * F + f;
    float val = (idx[sidx] == j) ? src[sidx] : 0.f;
    dst[((size_t)v * BATCH + b) * Fdst + off + f] = val;
}

// copy into concat dst
__global__ void k_copy_cat(float* __restrict__ dst, int Fdst, int off,
                           const float* __restrict__ src, int V, int F) {
    int t = blockIdx.x * blockDim.x + threadIdx.x;
    if (t >= V * BATCH * F) return;
    int f = t % F; int vb = t / F;
    dst[(size_t)vb * Fdst + off + f] = src[t];
}

// y [V,B,F] -> out [B,V,F]
__global__ void k_out(float* __restrict__ out, const float* __restrict__ y, int V, int F) {
    int t = blockIdx.x * blockDim.x + threadIdx.x;
    if (t >= V * BATCH * F) return;
    int f = t % F; int vb = t / F; int b = vb % BATCH; int v = vb / BATCH;
    out[((size_t)b * V + v) * F + f] = y[t];
}

// ---------------------------------------------------------------------------
// Host orchestration
// ---------------------------------------------------------------------------
extern "C" void kernel_launch(void* const* d_in, const int* in_sizes, int n_in,
                              void* d_out, int out_size, void* d_ws, size_t ws_size,
                              hipStream_t stream) {
    (void)in_sizes; (void)n_in; (void)out_size; (void)ws_size;
    const int V0 = 3072, V1 = 768, V2 = 192;

    // ---- input pointer map (pytree: dicts flattened alphabetically) ----
    const float* x = (const float*)d_in[0];
    // alphabetical layer order -> leaf start index (b,[beta,gamma],w)
    static const int pstart[14] = {1,5,9,13,17,21,25,29,33,37,41,45,47,51};
    // laplacians after 54 param leaves (x at 0): 55..63
    const int* l0_col = (const int*)d_in[56]; const float* l0_val = (const float*)d_in[57];
    const int* l1_col = (const int*)d_in[59]; const float* l1_val = (const float*)d_in[60];
    const int* l2_col = (const int*)d_in[62]; const float* l2_val = (const float*)d_in[63];

    // ---- workspace carve-up (256B aligned) ----
    size_t off = 0;
    auto alloc = [&](size_t bytes) -> size_t {
        size_t o = off; off = (off + bytes + 255) & ~(size_t)255; return o;
    };
    char* ws = (char*)d_ws;
    const size_t M0 = (size_t)V0 * BATCH;                 // 49152
    float*  ACT_A = (float*) (ws + alloc(M0 * 128 * 4));
    float*  ACT_B = (float*) (ws + alloc(M0 * 128 * 4));
    float*  SKIP1 = (float*) (ws + alloc(M0 * 64 * 4));
    float*  SKIP2 = (float*) (ws + alloc((size_t)V1 * BATCH * 128 * 4));
    int*    IDX1  = (int*)   (ws + alloc((size_t)V1 * BATCH * 64 * 4));
    int*    IDX2  = (int*)   (ws + alloc((size_t)V2 * BATCH * 128 * 4));
    float*  T1    = (float*) (ws + alloc(M0 * 128 * 4));
    unsigned short* XK = (unsigned short*)(ws + alloc(M0 * 384 * 2));
    unsigned short* WT = (unsigned short*)(ws + alloc((size_t)256 * 768 * 2));
    float*  STATS = (float*) (ws + alloc(512 * 4));

    auto cdiv = [](int a, int b) { return (a + b - 1) / b; };

    // one ConvCheb (+ optional BN/ReLU), src -> dst
    auto conv = [&](const float* src, float* dst, int V, int fin, int fout,
                    const int* col, const float* val, int alphaIdx, bool bn) {
        int base = pstart[alphaIdx];
        const float* bias  = (const float*)d_in[base + 0];
        const float* beta  = bn ? (const float*)d_in[base + 1] : nullptr;
        const float* gamma = bn ? (const float*)d_in[base + 2] : nullptr;
        const float* w     = (const float*)d_in[base + (bn ? 3 : 1)];

        int Kd = 3 * fin;
        int Kpad = ((Kd + 31) / 32) * 32;
        int Npad = ((fout + 15) / 16) * 16;
        int M = V * BATCH;

        int nw = Npad * Kpad;
        k_prepw<<<cdiv(nw, 256), 256, 0, stream>>>(WT, w, fin, fout, Kd, Kpad, Npad);

        int nc = V * BATCH * fin;
        k_cheb1<<<cdiv(nc, 256), 256, 0, stream>>>(XK, T1, src, col, val, V, fin, Kpad);
        if (Kpad > Kd) {
            int np = M * (Kpad - Kd);
            k_zpad<<<cdiv(np, 256), 256, 0, stream>>>(XK, Kd, Kpad, M);
        }
        k_cheb2<<<cdiv(nc, 256), 256, 0, stream>>>(XK, T1, src, col, val, V, fin, Kpad);

        int tilesM = M / 16, tilesN = Npad / 16;
        int tiles = tilesM * tilesN;
        k_gemm<<<cdiv(tiles, 4), 128, 0, stream>>>(dst, XK, WT, bias, Kpad, fout, tilesM, tilesN);

        if (bn) {
            k_stats<<<fout, 256, 0, stream>>>(STATS, dst, M, fout);
            int ne = M * fout;
            k_bnrelu<<<cdiv(ne, 256), 256, 0, stream>>>(dst, STATS, gamma, beta, M, fout);
        }
    };

    // ---- forward pass ----
    {
        int n = V0 * BATCH * 8;
        k_transpose_in<<<cdiv(n, 256), 256, 0, stream>>>(ACT_A, x, V0, 8);
    }
    conv(ACT_A, ACT_B, V0,   8,  16, l0_col, l0_val, 0, true);   // conv11
    conv(ACT_B, ACT_A, V0,  16,  32, l0_col, l0_val, 1, true);   // conv12
    conv(ACT_A, SKIP1, V0,  32,  64, l0_col, l0_val, 2, true);   // conv13 -> x1

    { int n = V1 * BATCH * 64;
      k_pool<<<cdiv(n, 256), 256, 0, stream>>>(ACT_A, IDX1, SKIP1, V1, 64); }

    conv(ACT_A, ACT_B, V1,  64,  88, l1_col, l1_val, 3, true);   // conv21
    conv(ACT_B, ACT_A, V1,  88, 110, l1_col, l1_val, 4, true);   // conv22
    conv(ACT_A, SKIP2, V1, 110, 128, l1_col, l1_val, 5, true);   // conv23 -> x2

    { int n = V2 * BATCH * 128;
      k_pool<<<cdiv(n, 256), 256, 0, stream>>>(ACT_A, IDX2, SKIP2, V2, 128); }

    conv(ACT_A, ACT_B, V2, 128, 256, l2_col, l2_val, 6, true);   // conv31
    conv(ACT_B, ACT_A, V2, 256, 256, l2_col, l2_val, 7, true);   // conv32
    conv(ACT_A, ACT_B, V2, 256, 128, l2_col, l2_val, 8, true);   // conv33 -> x3

    // d1 = concat(unpool(x3, idx2), x2)  -> ACT_A [V1, B, 256]
    { int n = V1 * BATCH * 128;
      k_unpool_cat<<<cdiv(n, 256), 256, 0, stream>>>(ACT_A, 256, 0, ACT_B, IDX2, V2, 128);
      k_copy_cat<<<cdiv(n, 256), 256, 0, stream>>>(ACT_A, 256, 128, SKIP2, V1, 128); }

    conv(ACT_A, ACT_B, V1, 256, 128, l1_col, l1_val, 12, true);  // uconv21
    conv(ACT_B, ACT_A, V1, 128,  64, l1_col, l1_val, 13, true);  // uconv22

    // d0 = concat(unpool(d, idx1), x1)  -> ACT_B [V0, B, 128]
    { int n = V0 * BATCH * 64;
      k_unpool_cat<<<cdiv(n, 256), 256, 0, stream>>>(ACT_B, 128, 0, ACT_A, IDX1, V1, 64);
      k_copy_cat<<<cdiv(n, 256), 256, 0, stream>>>(ACT_B, 128, 64, SKIP1, V0, 64); }

    conv(ACT_B, ACT_A, V0, 128,  64, l0_col, l0_val,  9, true);  // uconv11
    conv(ACT_A, ACT_B, V0,  64,  32, l0_col, l0_val, 10, true);  // uconv12
    conv(ACT_B, ACT_A, V0,  32,   2, l0_col, l0_val, 11, false); // uconv13 (no BN)

    { int n = V0 * BATCH * 2;
      k_out<<<cdiv(n, 256), 256, 0, stream>>>((float*)d_out, ACT_A, V0, 2); }
}
